// GPT_27822798143852
// MI455X (gfx1250) — matrix operand order repaired and checked by reading, not compiled, not run on hardware
//
#include <hip/hip_runtime.h>
#include <math.h>

// ---------------- types ----------------
typedef __bf16 bf16;
typedef __attribute__((ext_vector_type(16))) __bf16 v16bf;
typedef __attribute__((ext_vector_type(8)))  __bf16 v8bf;
typedef __attribute__((ext_vector_type(8)))  float   v8f;

#define D_MODEL 1024
#define N_HEAD  16
#define HEAD_D  64
#define SEQ     1024
#define BATCH   4
#define NLAYER  6
#define FFDIM   4096
#define MROWS   4096          // B*S tokens
#define VOCAB   50257

static __device__ __forceinline__ v16bf cat8(v8bf lo, v8bf hi) {
  return __builtin_shufflevector(lo, hi, 0,1,2,3,4,5,6,7,8,9,10,11,12,13,14,15);
}

// LDS byte offset of a __shared__ object: the flat (generic) address of an LDS
// location keeps the wave-relative LDS offset in its low 32 bits.
static __device__ __forceinline__ unsigned lds_off(const void* p) {
  return (unsigned)(unsigned long long)(uintptr_t)p;
}

// async global -> LDS 16B copy (CDNA5; tracked by ASYNCcnt)
static __device__ __forceinline__ void async_copy16(unsigned ldst, const void* gsrc) {
  asm volatile("global_load_async_to_lds_b128 %0, %1, off"
               :: "v"(ldst), "v"((unsigned long long)(uintptr_t)gsrc)
               : "memory");
}
static __device__ __forceinline__ void async_wait0() {
  asm volatile("s_wait_asynccnt 0x0" ::: "memory");
}

// ---------------- embedding: x = tok_emb[idx] + pos_emb ----------------
__global__ __launch_bounds__(256) void embed_kernel(const int* __restrict__ idx,
                                                    const float* __restrict__ tok,
                                                    const float* __restrict__ pos,
                                                    float* __restrict__ x) {
  const int row = blockIdx.x;            // b*S + s
  const int token = idx[row];
  const int s = row & (SEQ - 1);
  const int t = threadIdx.x;
#pragma unroll
  for (int i = 0; i < 4; ++i) {
    const int c = t + 256 * i;
    x[(size_t)row * D_MODEL + c] =
        tok[(size_t)token * D_MODEL + c] + pos[(size_t)s * D_MODEL + c];
  }
}

// ---------------- layernorm (f32 in -> bf16 out) ----------------
__global__ __launch_bounds__(256) void ln_kernel(const float* __restrict__ x,
                                                 const float* __restrict__ g,
                                                 const float* __restrict__ sh,
                                                 bf16* __restrict__ out) {
  const int row = blockIdx.x;
  const float* xr = x + (size_t)row * D_MODEL;
  const int t = threadIdx.x;
  float v[4], s = 0.f, ss = 0.f;
#pragma unroll
  for (int i = 0; i < 4; ++i) {
    v[i] = xr[t + 256 * i];
    s += v[i];
    ss += v[i] * v[i];
  }
#pragma unroll
  for (int off = 16; off >= 1; off >>= 1) {
    s  += __shfl_xor(s,  off);
    ss += __shfl_xor(ss, off);
  }
  __shared__ float sm[8], sm2[8];
  const int wave = t >> 5;
  if ((t & 31) == 0) { sm[wave] = s; sm2[wave] = ss; }
  __syncthreads();
  s = 0.f; ss = 0.f;
#pragma unroll
  for (int w = 0; w < 8; ++w) { s += sm[w]; ss += sm2[w]; }
  const float mean = s * (1.f / D_MODEL);
  const float var  = ss * (1.f / D_MODEL) - mean * mean;
  const float rstd = rsqrtf(var + 1e-5f);
  bf16* orow = out + (size_t)row * D_MODEL;
#pragma unroll
  for (int i = 0; i < 4; ++i) {
    const int c = t + 256 * i;
    orow[c] = (bf16)((v[i] - mean) * rstd * g[c] + sh[c]);
  }
}

// ---------------- GEMM: C[M,N] = A_bf16[M,K] @ W_f32[N,K]^T (+bias, fused epilogue)
// MODE 0: f32 store            MODE 1: f32 store + residual add (resid)
// MODE 2: bf16 store           MODE 3: bf16 store of exact GELU
// BOUND: guard N edge (only lm_head needs it).
// block = 256 threads (8 waves), tile 128(M) x 64(N), BK=32, wave tile 32x32.
// Ping-pong LDS, A staged via async global->LDS, B staged via f32 loads + cvt.
template <int MODE, bool BOUND>
__global__ __launch_bounds__(256) void gemm_kernel(const bf16* __restrict__ A,
                                                   const float* __restrict__ W,
                                                   const float* __restrict__ bias,
                                                   const float* __restrict__ resid,
                                                   float* __restrict__ outF,
                                                   bf16* __restrict__ outB,
                                                   int N, int K) {
  __shared__ __align__(16) bf16 As[2][128 * 40];  // 128 rows x 32 K, stride 40
  __shared__ __align__(16) bf16 Bs[2][64 * 40];   // 64 n-rows x 32 K, stride 40
  const int t    = threadIdx.x;
  const int lane = t & 31;
  const int wave = t >> 5;
  const int half = lane >> 4;
  const int l16  = lane & 15;
  const int wm   = wave & 3;     // 4 row groups of 32
  const int wn   = wave >> 2;    // 2 col groups of 32
  const int m0 = blockIdx.y * 128;
  const int n0 = blockIdx.x * 64;

  // per-thread staging coordinates (512 x 16B chunks for A, 512 x 4-float for B)
  const int rA0 = (t * 2) >> 2,     cA0 = ((t * 2) & 3) * 8;
  const int rA1 = (t * 2 + 1) >> 2, cA1 = ((t * 2 + 1) & 3) * 8;
  const int nB0 = (t * 2) >> 3,     cB0 = ((t * 2) & 7) * 4;
  const int nB1 = (t * 2 + 1) >> 3, cB1 = ((t * 2 + 1) & 7) * 4;
  const bool okB0 = !BOUND || (n0 + nB0 < N);
  const bool okB1 = !BOUND || (n0 + nB1 < N);

  const int nK = K >> 5;
  v8f acc[2][2] = {};

  // ---- prologue: stage tile 0 into buffer 0 ----
  {
    async_copy16(lds_off(&As[0][rA0 * 40 + cA0]),
                 A + (size_t)(m0 + rA0) * K + cA0);
    async_copy16(lds_off(&As[0][rA1 * 40 + cA1]),
                 A + (size_t)(m0 + rA1) * K + cA1);
    float4 f0 = make_float4(0.f, 0.f, 0.f, 0.f), f1 = f0;
    if (okB0) f0 = *(const float4*)(W + (size_t)(n0 + nB0) * K + cB0);
    if (okB1) f1 = *(const float4*)(W + (size_t)(n0 + nB1) * K + cB1);
    bf16* d0 = &Bs[0][nB0 * 40 + cB0];
    d0[0] = (bf16)f0.x; d0[1] = (bf16)f0.y; d0[2] = (bf16)f0.z; d0[3] = (bf16)f0.w;
    bf16* d1 = &Bs[0][nB1 * 40 + cB1];
    d1[0] = (bf16)f1.x; d1[1] = (bf16)f1.y; d1[2] = (bf16)f1.z; d1[3] = (bf16)f1.w;
    async_wait0();
  }
  __syncthreads();

  int buf = 0;
  for (int kk = 0; kk < nK; ++kk) {
    const bool more = (kk + 1) < nK;
    float4 f0 = make_float4(0.f, 0.f, 0.f, 0.f), f1 = f0;
    if (more) {
      const int k1 = (kk + 1) * 32;
      const int nb = buf ^ 1;
      async_copy16(lds_off(&As[nb][rA0 * 40 + cA0]),
                   A + (size_t)(m0 + rA0) * K + k1 + cA0);
      async_copy16(lds_off(&As[nb][rA1 * 40 + cA1]),
                   A + (size_t)(m0 + rA1) * K + k1 + cA1);
      if (okB0) f0 = *(const float4*)(W + (size_t)(n0 + nB0) * K + k1 + cB0);
      if (okB1) f1 = *(const float4*)(W + (size_t)(n0 + nB1) * K + k1 + cB1);
    }

    // ---- compute current buffer ----
    v16bf afr[2], bfr[2];
#pragma unroll
    for (int i = 0; i < 2; ++i) {
      const bf16* p = &As[buf][(wm * 32 + i * 16 + l16) * 40];
      afr[i] = cat8(*(const v8bf*)(p + 8 * half),
                    *(const v8bf*)(p + 16 + 8 * half));
    }
#pragma unroll
    for (int j = 0; j < 2; ++j) {
      const bf16* p = &Bs[buf][(wn * 32 + j * 16 + l16) * 40 + 8 * half];
      bfr[j] = cat8(*(const v8bf*)p, *(const v8bf*)(p + 16));
    }
#pragma unroll
    for (int i = 0; i < 2; ++i)
#pragma unroll
      for (int j = 0; j < 2; ++j)
        acc[i][j] = __builtin_amdgcn_wmma_f32_16x16x32_bf16(
            false, afr[i], false, bfr[j], (short)0, acc[i][j], false, false);

    if (more) {
      const int nb = buf ^ 1;
      bf16* d0 = &Bs[nb][nB0 * 40 + cB0];
      d0[0] = (bf16)f0.x; d0[1] = (bf16)f0.y; d0[2] = (bf16)f0.z; d0[3] = (bf16)f0.w;
      bf16* d1 = &Bs[nb][nB1 * 40 + cB1];
      d1[0] = (bf16)f1.x; d1[1] = (bf16)f1.y; d1[2] = (bf16)f1.z; d1[3] = (bf16)f1.w;
      async_wait0();
    }
    __syncthreads();
    buf ^= 1;
  }

  // ---- fused epilogue ----
#pragma unroll
  for (int i = 0; i < 2; ++i)
#pragma unroll
    for (int j = 0; j < 2; ++j) {
      const int n = n0 + wn * 32 + j * 16 + l16;
      if (BOUND && n >= N) continue;
      const float bv = bias ? bias[n] : 0.f;
#pragma unroll
      for (int v = 0; v < 8; ++v) {
        const int mrow = m0 + wm * 32 + i * 16 + v + 8 * half;
        const size_t off = (size_t)mrow * N + n;
        const float val = acc[i][j][v] + bv;
        if constexpr (MODE == 0)      outF[off] = val;
        else if constexpr (MODE == 1) outF[off] = resid[off] + val;
        else if constexpr (MODE == 2) outB[off] = (bf16)val;
        else                          outB[off] = (bf16)(0.5f * val *
                                          (1.f + erff(val * 0.70710678118654752f)));
      }
    }
}

// ---------------- flash attention: one (b, h, 64-query tile) per block ----------------
// 128 threads = 4 waves, each wave owns 16 query rows (online softmax).
__global__ __launch_bounds__(128) void attn_kernel(const bf16* __restrict__ qkv,
                                                   bf16* __restrict__ ctx) {
  const int q0 = blockIdx.x * 64;
  const int bh = blockIdx.y;
  const int b  = bh >> 4;           // H = 16
  const int h  = bh & 15;
  const int t    = threadIdx.x;
  const int lane = t & 31;
  const int wave = t >> 5;
  const int half = lane >> 4;
  const int l16  = lane & 15;
  const size_t rowS = 3 * D_MODEL;
  const size_t base = (size_t)b * SEQ;

  __shared__ __align__(16) bf16 Vt[64][72];      // V transposed: [hd][key]
  __shared__ __align__(16) bf16 Ps[4][16][72];   // per-wave P tile [row][key]

  // Q fragments (A layout): row = q0 + wave*16 + l16, K-dim = head dim (64)
  const bf16* qrow = qkv + (base + q0 + wave * 16 + l16) * rowS + h * HEAD_D;
  v16bf aq[2];
#pragma unroll
  for (int kb = 0; kb < 2; ++kb)
    aq[kb] = cat8(*(const v8bf*)(qrow + kb * 32 + 8 * half),
                  *(const v8bf*)(qrow + kb * 32 + 16 + 8 * half));

  v8f o[4] = {};
  float m[8], lsum[8];
#pragma unroll
  for (int v = 0; v < 8; ++v) { m[v] = -1e30f; lsum[v] = 0.f; }

  for (int kt = 0; kt <= (int)blockIdx.x; ++kt) {
    const int k0 = kt * 64;
    __syncthreads();
    // stage V tile transposed into LDS (coalesced reads along hd)
    for (int i = t; i < 64 * 64; i += 128) {
      const int key = i >> 6, hd = i & 63;
      Vt[hd][key] =
          qkv[(base + k0 + key) * rowS + 2 * D_MODEL + h * HEAD_D + hd];
    }
    __syncthreads();

    // S = Q @ K^T : B fragments come straight from K rows (contiguous in hd)
    v8f sc[4] = {};
#pragma unroll
    for (int nt = 0; nt < 4; ++nt) {
      const bf16* krow =
          qkv + (base + k0 + nt * 16 + l16) * rowS + D_MODEL + h * HEAD_D;
#pragma unroll
      for (int kb = 0; kb < 2; ++kb) {
        const v16bf bk = cat8(*(const v8bf*)(krow + kb * 32 + 8 * half),
                              *(const v8bf*)(krow + kb * 32 + 16 + 8 * half));
        sc[nt] = __builtin_amdgcn_wmma_f32_16x16x32_bf16(
            false, aq[kb], false, bk, (short)0, sc[nt], false, false);
      }
    }

    // scale, causal mask, row max
    float rm[8];
#pragma unroll
    for (int v = 0; v < 8; ++v) {
      const int rowg = q0 + wave * 16 + v + 8 * half;
      float mx = -1e30f;
#pragma unroll
      for (int nt = 0; nt < 4; ++nt) {
        const int colg = k0 + nt * 16 + l16;
        float sv = sc[nt][v] * 0.125f;          // 1/sqrt(64)
        sv = (colg > rowg) ? -1e30f : sv;
        sc[nt][v] = sv;
        mx = fmaxf(mx, sv);
      }
      rm[v] = mx;
    }
#pragma unroll
    for (int off = 1; off < 16; off <<= 1)
#pragma unroll
      for (int v = 0; v < 8; ++v)
        rm[v] = fmaxf(rm[v], __shfl_xor(rm[v], off));

    // online softmax update
    float corr[8], pls[8];
#pragma unroll
    for (int v = 0; v < 8; ++v) {
      const float mnew = fmaxf(m[v], rm[v]);
      corr[v] = __expf(m[v] - mnew);
      m[v] = mnew;
      float ps = 0.f;
#pragma unroll
      for (int nt = 0; nt < 4; ++nt) {
        const float p = __expf(sc[nt][v] - mnew);
        sc[nt][v] = p;
        ps += p;
      }
      pls[v] = ps;
    }
#pragma unroll
    for (int off = 1; off < 16; off <<= 1)
#pragma unroll
      for (int v = 0; v < 8; ++v)
        pls[v] += __shfl_xor(pls[v], off);
#pragma unroll
    for (int v = 0; v < 8; ++v) {
      lsum[v] = lsum[v] * corr[v] + pls[v];
#pragma unroll
      for (int nt = 0; nt < 4; ++nt)
        o[nt][v] *= corr[v];
    }

    // P: C layout -> LDS (row major) so it can be reloaded in A layout
#pragma unroll
    for (int v = 0; v < 8; ++v)
#pragma unroll
      for (int nt = 0; nt < 4; ++nt)
        Ps[wave][v + 8 * half][nt * 16 + l16] = (bf16)sc[nt][v];

    // O += P @ V  (K-dim = 64 keys)
#pragma unroll
    for (int kb = 0; kb < 2; ++kb) {
      const bf16* prow = &Ps[wave][l16][0];
      const v16bf ap = cat8(*(const v8bf*)(prow + kb * 32 + 8 * half),
                            *(const v8bf*)(prow + kb * 32 + 16 + 8 * half));
#pragma unroll
      for (int nt = 0; nt < 4; ++nt) {
        const bf16* vrow = &Vt[nt * 16 + l16][kb * 32 + 8 * half];
        const v16bf bv = cat8(*(const v8bf*)vrow, *(const v8bf*)(vrow + 16));
        o[nt] = __builtin_amdgcn_wmma_f32_16x16x32_bf16(
            false, ap, false, bv, (short)0, o[nt], false, false);
      }
    }
  }

  // finalize: ctx[b, q, h*64 + hd] = O / l
#pragma unroll
  for (int v = 0; v < 8; ++v) {
    const float inv = 1.f / lsum[v];
    const size_t orow =
        (base + q0 + wave * 16 + v + 8 * half) * D_MODEL + h * HEAD_D;
#pragma unroll
    for (int nt = 0; nt < 4; ++nt)
      ctx[orow + nt * 16 + l16] = (bf16)(o[nt][v] * inv);
  }
}

// ---------------- host driver ----------------
extern "C" void kernel_launch(void* const* d_in, const int* in_sizes, int n_in,
                              void* d_out, int out_size, void* d_ws, size_t ws_size,
                              hipStream_t stream) {
  const int*   idx   = (const int*)  d_in[0];
  const float* tok   = (const float*)d_in[1];
  const float* pos   = (const float*)d_in[2];
  const float* ln1s  = (const float*)d_in[3];
  const float* ln1b  = (const float*)d_in[4];
  const float* attnw = (const float*)d_in[5];
  const float* attnb = (const float*)d_in[6];
  const float* projw = (const float*)d_in[7];
  const float* projb = (const float*)d_in[8];
  const float* ln2s  = (const float*)d_in[9];
  const float* ln2b  = (const float*)d_in[10];
  const float* fc1w  = (const float*)d_in[11];
  const float* fc1b  = (const float*)d_in[12];
  const float* fc2w  = (const float*)d_in[13];
  const float* fc2b  = (const float*)d_in[14];
  const float* lnfs  = (const float*)d_in[15];
  const float* lnfb  = (const float*)d_in[16];

  char* wsp = (char*)d_ws;
  float* x   = (float*)wsp; wsp += (size_t)MROWS * D_MODEL * 4;     // 16 MB
  bf16*  xb  = (bf16*)wsp;  wsp += (size_t)MROWS * D_MODEL * 2;     //  8 MB
  bf16*  qkv = (bf16*)wsp;  wsp += (size_t)MROWS * 3 * D_MODEL * 2; // 24 MB
  bf16*  ctx = (bf16*)wsp;  wsp += (size_t)MROWS * D_MODEL * 2;     //  8 MB
  bf16*  ff  = (bf16*)wsp;  wsp += (size_t)MROWS * FFDIM * 2;       // 32 MB

  const dim3 gQKV((3 * D_MODEL) / 64, MROWS / 128);
  const dim3 gD(D_MODEL / 64, MROWS / 128);
  const dim3 gFF(FFDIM / 64, MROWS / 128);
  const dim3 gLM((VOCAB + 63) / 64, MROWS / 128);
  const dim3 gAT(SEQ / 64, BATCH * N_HEAD);

  embed_kernel<<<MROWS, 256, 0, stream>>>(idx, tok, pos, x);
  for (int l = 0; l < NLAYER; ++l) {
    ln_kernel<<<MROWS, 256, 0, stream>>>(x, ln1s + l * D_MODEL,
                                         ln1b + l * D_MODEL, xb);
    gemm_kernel<2, false><<<gQKV, 256, 0, stream>>>(
        xb, attnw + (size_t)l * 3 * D_MODEL * D_MODEL,
        attnb + (size_t)l * 3 * D_MODEL, nullptr, nullptr, qkv,
        3 * D_MODEL, D_MODEL);
    attn_kernel<<<gAT, 128, 0, stream>>>(qkv, ctx);
    gemm_kernel<1, false><<<gD, 256, 0, stream>>>(
        ctx, projw + (size_t)l * D_MODEL * D_MODEL, projb + (size_t)l * D_MODEL,
        x, x, nullptr, D_MODEL, D_MODEL);
    ln_kernel<<<MROWS, 256, 0, stream>>>(x, ln2s + l * D_MODEL,
                                         ln2b + l * D_MODEL, xb);
    gemm_kernel<3, false><<<gFF, 256, 0, stream>>>(
        xb, fc1w + (size_t)l * FFDIM * D_MODEL, fc1b + (size_t)l * FFDIM,
        nullptr, nullptr, ff, FFDIM, D_MODEL);
    gemm_kernel<1, false><<<gD, 256, 0, stream>>>(
        ff, fc2w + (size_t)l * D_MODEL * FFDIM, fc2b + (size_t)l * D_MODEL,
        x, x, nullptr, D_MODEL, FFDIM);
  }
  ln_kernel<<<MROWS, 256, 0, stream>>>(x, lnfs, lnfb, xb);
  // tied lm_head, no bias: logits f32 straight to d_out
  gemm_kernel<0, true><<<gLM, 256, 0, stream>>>(xb, tok, nullptr, nullptr,
                                                (float*)d_out, nullptr,
                                                VOCAB, D_MODEL);
}
